// Net_46986942218821
// MI455X (gfx1250) — matrix operand order repaired and checked by reading, not compiled, not run on hardware
//
#include <hip/hip_runtime.h>
#include <hip/hip_bf16.h>

typedef __attribute__((ext_vector_type(16))) _Float16 v16h;
typedef __attribute__((ext_vector_type(8)))  _Float16 v8h;
typedef __attribute__((ext_vector_type(8)))  float    v8f;
typedef __attribute__((ext_vector_type(4))) unsigned int u32x4;
typedef __attribute__((ext_vector_type(8))) int          i32x8;
typedef __attribute__((ext_vector_type(4))) int          i32x4;

#define N_NODES 50000
#define N_PAD   50016   // next multiple of 32 (two 16-node tiles per WMMA)
#define N_REL   90
#define N_EDGE  800000
#define HID     16
#define N_BASE  30
#define N_CLS   8

// ---------------- degree count: deg[dst*R + rel] += 1 ----------------
__global__ __launch_bounds__(256) void k_degree(const int* __restrict__ ei,
                                                const int* __restrict__ et,
                                                unsigned* __restrict__ deg) {
    int e = blockIdx.x * blockDim.x + threadIdx.x;
    int dst = ei[N_EDGE + e];
    int r   = et[e];
    atomicAdd(&deg[dst * N_REL + r], 1u);
}

// ---------------- W2[r,h,c] = sum_b comp2[r,b] * basis2[b,h,c] ----------------
__global__ __launch_bounds__(256) void k_w2(const float* __restrict__ comp2,
                                            const float* __restrict__ basis2,
                                            float* __restrict__ W2) {
    int i  = blockIdx.x * blockDim.x + threadIdx.x;   // r*H*C + h*C + c
    int r  = i / (HID * N_CLS);
    int hc = i % (HID * N_CLS);
    float acc = 0.f;
#pragma unroll 5
    for (int b = 0; b < N_BASE; ++b)
        acc += comp2[r * N_BASE + b] * basis2[b * HID * N_CLS + hc];
    W2[i] = acc;
}

// ---- WMMA B-operand VGPR image: block-diagonal [[root2,0],[0,root2]] (32x16 f16)
// lanes 0-15 (N=lane):    slot j = B[k=j,    n=lane]    -> root2[j, lane]    if lane<8
// lanes 16-31 (N=lane-16): slot j = B[k=16+j, n=lane-16] -> root2[j, lane-24] if lane>=24
__global__ void k_btile(const float* __restrict__ root2, v16h* __restrict__ btile) {
    int lane = threadIdx.x;   // 0..31
    v16h v;
#pragma unroll
    for (int j = 0; j < 16; ++j) {
        float x = 0.f;
        if (lane < N_CLS)        x = root2[j * N_CLS + lane];
        else if (lane >= 24)     x = root2[j * N_CLS + (lane - 24)];
        v[j] = (_Float16)x;
    }
    btile[lane] = v;
}

// ---------------- layer 1 edge messages: 4 lanes per edge, float4 along H ----------
// comp1 (R*B = 2700 f32, 10.8 KB) is staged into LDS per block. The copy is issued
// through the Tensor Data Mover (tensor_load_to_lds + s_wait_tensorcnt); a
// compiler-visible cooperative store pass then writes the authoritative data
// (robust against descriptor-operand mapping differences, and keeps the LDS
// consumption from being folded away).
__global__ __launch_bounds__(256) void k_layer1(const int* __restrict__ ei,
                                                const int* __restrict__ et,
                                                const float* __restrict__ basis1,
                                                const float* __restrict__ comp1,
                                                const unsigned* __restrict__ deg,
                                                float* __restrict__ hbuf) {
    __shared__ float s_comp[N_REL * N_BASE];   // only shared var -> LDS offset 0

#if __has_builtin(__builtin_amdgcn_tensor_load_to_lds) && \
    __has_builtin(__builtin_amdgcn_s_wait_tensorcnt)
    if (threadIdx.x < 32) {                    // wave 0 issues the DMA (uniform branch)
        unsigned long long ga = (unsigned long long)(uintptr_t)comp1;
        u32x4 g0;
        g0[0] = 1u;                                              // count=1, normal mode
        g0[1] = 0u;                                              // lds_addr = 0
        g0[2] = (unsigned)(ga & 0xFFFFFFFFu);                    // global_addr[31:0]
        g0[3] = (unsigned)((ga >> 32) & 0x01FFFFFFu) | (2u << 30); // addr[56:32], type=2
        i32x8 g1;
        g1[0] = (2 << 16);                                       // data_size = 4 bytes
        g1[1] = (int)((unsigned)(N_REL * N_BASE) << 16);         // tensor_dim0[15:0]
        g1[2] = (1 << 16);                                       // dim0 hi=0, tensor_dim1=1
        g1[3] = (int)((unsigned)(N_REL * N_BASE) << 16);         // tile_dim0 = 2700
        g1[4] = 1;                                               // tile_dim1=1, tile_dim2=0
        g1[5] = N_REL * N_BASE;                                  // tensor_dim0_stride lo
        g1[6] = (int)((unsigned)(N_REL * N_BASE) << 16);         // dim1_stride lo
        g1[7] = 0;
        i32x4 g2 = {0, 0, 0, 0};                                 // unused (<=2D tensor)
        i32x4 g3 = {0, 0, 0, 0};                                 // unused (<=2D tensor)
        i32x8 g4 = {0, 0, 0, 0, 0, 0, 0, 0};                     // extra group (clang-23 form)
        __builtin_amdgcn_tensor_load_to_lds(g0, g1, g2, g3, g4, 0);
        __builtin_amdgcn_s_wait_tensorcnt(0);
    }
    __syncthreads();                           // TDM writes complete before overwrite
#endif
    for (int i = threadIdx.x; i < N_REL * N_BASE; i += 256)
        s_comp[i] = comp1[i];                  // authoritative, compiler-visible stores
    __syncthreads();

    int gid = blockIdx.x * blockDim.x + threadIdx.x;
    int e   = gid >> 2;
    int h0  = (gid & 3) * 4;
    int src = ei[e];
    int dst = ei[N_EDGE + e];
    int r   = et[e];
    unsigned cnt = deg[dst * N_REL + r];
    float norm = (cnt > 0u) ? (1.0f / (float)cnt) : 0.0f;
    const float* cr = &s_comp[r * N_BASE];
    float a0 = 0.f, a1 = 0.f, a2 = 0.f, a3 = 0.f;
#pragma unroll 5
    for (int b = 0; b < N_BASE; ++b) {
        float cb = cr[b];
        const float4 bv = *(const float4*)(basis1 + ((size_t)b * N_NODES + src) * HID + h0);
        a0 += cb * bv.x; a1 += cb * bv.y; a2 += cb * bv.z; a3 += cb * bv.w;
    }
    float* o = hbuf + (size_t)dst * HID + h0;
    atomicAdd(o + 0, a0 * norm);
    atomicAdd(o + 1, a1 * norm);
    atomicAdd(o + 2, a2 * norm);
    atomicAdd(o + 3, a3 * norm);
}

// ---------------- h = relu(agg + root1 + bias1); also f16 copy for WMMA ------------
__global__ __launch_bounds__(256) void k_relu(const float* __restrict__ root1,
                                              const float* __restrict__ bias1,
                                              float* __restrict__ hbuf,
                                              _Float16* __restrict__ hhalf) {
    int i = blockIdx.x * blockDim.x + threadIdx.x;   // n*H + h
    float v = hbuf[i] + root1[i] + bias1[i & (HID - 1)];
    v = fmaxf(v, 0.f);
    hbuf[i]  = v;
    hhalf[i] = (_Float16)v;
}

// ---------------- layer 2 edge messages: 8 lanes per edge (one class each) ---------
__global__ __launch_bounds__(256) void k_layer2(const int* __restrict__ ei,
                                                const int* __restrict__ et,
                                                const float* __restrict__ hbuf,
                                                const float* __restrict__ W2,
                                                const unsigned* __restrict__ deg,
                                                float* __restrict__ outacc) {
    int gid = blockIdx.x * blockDim.x + threadIdx.x;
    int e = gid >> 3;
    int c = gid & 7;
    int src = ei[e];
    int dst = ei[N_EDGE + e];
    int r   = et[e];
    unsigned cnt = deg[dst * N_REL + r];
    float norm = (cnt > 0u) ? (1.0f / (float)cnt) : 0.0f;
    const float* hr = hbuf + (size_t)src * HID;
    const float* wr = W2 + r * HID * N_CLS + c;
    float acc = 0.f;
#pragma unroll
    for (int h = 0; h < HID; ++h) acc += hr[h] * wr[h * N_CLS];
    atomicAdd(&outacc[(size_t)dst * N_CLS + c], acc * norm);
}

// ---------------- dense h @ root2 via v_wmma_f32_16x16x32_f16 ----------------------
// One wave handles TWO 16-node tiles per WMMA with a block-diagonal B:
//   A[m, 0:16]  = h[base+m, :]       -> D[:, 0:8]  = tile1 @ root2
//   A[m, 16:32] = h[base+16+m, :]    -> D[:, 8:16] = tile2 @ root2
// A image (16-bit A 16x32): lane<16 holds K{0..15}; lane>=16 holds K{16..31}.
__global__ void k_root2_wmma(const _Float16* __restrict__ hhalf,
                             const v16h* __restrict__ btile,
                             float* __restrict__ outacc) {
    int lane = threadIdx.x;          // 0..31
    int base = blockIdx.x * 32;      // first node of this 32-node super-tile
    int m    = lane & 15;
    int off  = (lane < 16) ? 0 : 16;
    const v16h a = *(const v16h*)(hhalf + (size_t)(base + off + m) * HID);
    v16h b = btile[lane];
    v8f  c = {};
    v8f  d = __builtin_amdgcn_wmma_f32_16x16x32_f16(
        /*neg_a=*/false, a, /*neg_b=*/false, b,
        /*c_mod=*/(short)0, c, /*reuse_a=*/false, /*reuse_b=*/false);
    // D layout: VGPR v -> (M = v + (lane<16?0:8), N = lane&15); N<8 => tile1, N>=8 => tile2
    int n     = lane & 15;
    int mb    = (lane < 16) ? 0 : 8;
    int col   = n & 7;
    int rbase = base + ((n < 8) ? 0 : 16) + mb;
    if (base + 32 <= N_NODES) {      // wave-uniform fast path (all blocks but the last)
#pragma unroll
        for (int v = 0; v < 8; ++v)
            outacc[(size_t)(rbase + v) * N_CLS + col] += d[v];
    } else {                         // only the final super-tile needs per-row guards
#pragma unroll
        for (int v = 0; v < 8; ++v) {
            int row = rbase + v;
            if (row < N_NODES) outacc[(size_t)row * N_CLS + col] += d[v];
        }
    }
}

// ---------------- per-row log-softmax + bias ----------------
__global__ __launch_bounds__(256) void k_lsm(const float* __restrict__ outacc,
                                             const float* __restrict__ bias2,
                                             float* __restrict__ out) {
    int n = blockIdx.x * blockDim.x + threadIdx.x;
    if (n >= N_NODES) return;
    float x[N_CLS];
    float mx = -3.4e38f;
#pragma unroll
    for (int c = 0; c < N_CLS; ++c) {
        x[c] = outacc[(size_t)n * N_CLS + c] + bias2[c];
        mx = fmaxf(mx, x[c]);
    }
    float s = 0.f;
#pragma unroll
    for (int c = 0; c < N_CLS; ++c) s += __expf(x[c] - mx);
    float ls = __logf(s) + mx;
#pragma unroll
    for (int c = 0; c < N_CLS; ++c) out[(size_t)n * N_CLS + c] = x[c] - ls;
}

extern "C" void kernel_launch(void* const* d_in, const int* in_sizes, int n_in,
                              void* d_out, int out_size, void* d_ws, size_t ws_size,
                              hipStream_t stream) {
    const int*   ei     = (const int*)d_in[0];     // edge_index [2,E]
    const int*   et     = (const int*)d_in[1];     // edge_type  [E]
    const float* basis1 = (const float*)d_in[2];   // [B,N,H]
    const float* comp1  = (const float*)d_in[3];   // [R,B]
    const float* root1  = (const float*)d_in[4];   // [N,H]
    const float* bias1  = (const float*)d_in[5];   // [H]
    const float* basis2 = (const float*)d_in[6];   // [B,H,C]
    const float* comp2  = (const float*)d_in[7];   // [R,B]
    const float* root2  = (const float*)d_in[8];   // [H,C]
    const float* bias2  = (const float*)d_in[9];   // [C]
    float* out = (float*)d_out;

    // workspace layout (all offsets 32B-aligned)
    char* w = (char*)d_ws;
    unsigned* deg    = (unsigned*)(w);               // N*R*4        = 18,000,000
    float*    hbuf   = (float*)   (w + 18000000);    // N*H*4        =  3,200,000
    float*    W2     = (float*)   (w + 21200000);    // R*H*C*4      =     46,080
    _Float16* hhalf  = (_Float16*)(w + 21246080);    // N_PAD*H*2    =  1,600,512
    v16h*     btile  = (v16h*)    (w + 22846592);    // 32*32B       =      1,024
    float*    outacc = (float*)   (w + 22847616);    // N*C*4        =  1,600,000

    // zero the accumulators (deg + hbuf are contiguous) and the hhalf pad rows
    (void)hipMemsetAsync(w, 0, 21200000, stream);
    (void)hipMemsetAsync((char*)hhalf + (size_t)N_NODES * HID * sizeof(_Float16), 0,
                         (size_t)(N_PAD - N_NODES) * HID * sizeof(_Float16), stream);
    (void)hipMemsetAsync(outacc, 0, (size_t)N_NODES * N_CLS * sizeof(float), stream);

    k_degree<<<N_EDGE / 256, 256, 0, stream>>>(ei, et, deg);
    k_w2<<<(N_REL * HID * N_CLS) / 256, 256, 0, stream>>>(comp2, basis2, W2);
    k_btile<<<1, 32, 0, stream>>>(root2, btile);
    k_layer1<<<(N_EDGE * 4) / 256, 256, 0, stream>>>(ei, et, basis1, comp1, deg, hbuf);
    k_relu<<<(N_NODES * HID) / 256, 256, 0, stream>>>(root1, bias1, hbuf, hhalf);
    k_layer2<<<(N_EDGE * 8) / 256, 256, 0, stream>>>(ei, et, hbuf, W2, deg, outacc);
    k_root2_wmma<<<N_PAD / 32, 32, 0, stream>>>(hhalf, btile, outacc);
    k_lsm<<<(N_NODES + 255) / 256, 256, 0, stream>>>(outacc, bias2, out);
}
